// SelfAttention_77670188581159
// MI455X (gfx1250) — matrix operand order repaired and checked by reading, not compiled, not run on hardware
//
#include <hip/hip_runtime.h>
#include <hip/hip_bf16.h>

// ---------------------------------------------------------------------------
// MI455X (gfx1250) self-attention, bf16 WMMA everywhere.
//   B=16, S=1024, D=1024, H=16, Dh=64
// v_wmma_f32_16x16x32_bf16 with fp32 accumulation; software-pipelined tiles.
// ---------------------------------------------------------------------------

typedef __attribute__((ext_vector_type(16))) __bf16        v16bf;
typedef __attribute__((ext_vector_type(8)))  float         v8f;
typedef __attribute__((ext_vector_type(4)))  float         v4f;
typedef __attribute__((ext_vector_type(4)))  unsigned int  u32x4;

#define LOG2E 1.4426950408889634f

union FragBF {       // 16 bf16 = 8 VGPRs; loaded as two 16-byte LDS reads
  v16bf v;
  u32x4 q[2];
};

union RegTile {      // 16 bf16 staged in registers for LDS transpose stores
  u32x4  q[2];
  __bf16 h[16];
};

static __device__ __forceinline__ v8f wmma_bf16(const FragBF& a, const FragBF& b, v8f c) {
  return __builtin_amdgcn_wmma_f32_16x16x32_bf16(
      /*neg_a=*/false, a.v, /*neg_b=*/false, b.v,
      /*c_mod=*/(short)0, c, /*reuse_a=*/false, /*reuse_b=*/false);
}

// ---------------------------------------------------------------------------
// Kernel 1: Y = X(fp32) * W(fp32) + b  ->  bf16 out in [B,H,S,Dh] layout.
// Tile: 128(M) x 64(N), K-step 32. 8 waves; wave w owns rows 16w..16w+15.
// Software pipeline: global loads for k0+32 issued before compute of k0.
// ---------------------------------------------------------------------------
__global__ __launch_bounds__(256) void k_gemm_qkv(
    const float* __restrict__ X, const float* __restrict__ W,
    const float* __restrict__ bias, __bf16* __restrict__ Out) {
  constexpr int Dk = 1024, N = 1024, S = 1024, H = 16;
  __shared__ __bf16 Xs[128][40];  // 128x32 bf16 (+8 pad keeps 16B align)
  __shared__ __bf16 Ws[64][40];   // W tile TRANSPOSED: [n][k]
  const int tid  = threadIdx.x;
  const int lane = tid & 31, wave = tid >> 5;
  const int m0 = blockIdx.y * 128, n0 = blockIdx.x * 64;

  v8f acc[4] = {};

  const int xrow = tid >> 1, xcs = (tid & 1) * 16;    // X loader: 16 floats/thread
  const int wi = tid * 8, wk = wi >> 6, wn = wi & 63; // W loader: 8 floats/thread

  const int ar   = wave * 16 + (lane & 15);  // A-frag row (M = lane&15)
  const int kb   = (lane >> 4) * 8;          // A-frag K base (8-wide halves)
  const int bk   = (lane >> 4) * 8;          // B-frag K base (8-wide halves)
  const int lcol = lane & 15;

  v4f  xr[4];
  float wr[8];
  auto load_tiles = [&](int k0) {
    const float* xs = X + (size_t)(m0 + xrow) * Dk + k0 + xcs;
    #pragma unroll
    for (int j = 0; j < 4; ++j) xr[j] = *(const v4f*)(xs + 4 * j);
    const float* ws = W + (size_t)(k0 + wk) * N + n0 + wn;
    #pragma unroll
    for (int j = 0; j < 8; ++j) wr[j] = ws[j];
  };
  auto store_tiles = [&]() {
    #pragma unroll
    for (int j = 0; j < 4; ++j) {
      #pragma unroll
      for (int e = 0; e < 4; ++e) Xs[xrow][xcs + 4 * j + e] = (__bf16)xr[j][e];
    }
    #pragma unroll
    for (int j = 0; j < 8; ++j) Ws[wn + j][wk] = (__bf16)wr[j];
  };

  load_tiles(0);
  for (int k0 = 0; k0 < Dk; k0 += 32) {
    __syncthreads();               // prior compute done reading LDS
    store_tiles();
    __syncthreads();
    if (k0 + 32 < Dk) load_tiles(k0 + 32);  // hide global latency under WMMAs

    FragBF a;
    a.q[0] = *(const u32x4*)&Xs[ar][kb];        // K = kb..kb+7
    a.q[1] = *(const u32x4*)&Xs[ar][16 + kb];   // K = 16+kb..16+kb+7
    FragBF bfr[4];
    #pragma unroll
    for (int nb = 0; nb < 4; ++nb) {            // preload all B frags...
      const __bf16* wp = &Ws[nb * 16 + lcol][bk];
      bfr[nb].q[0] = *(const u32x4*)wp;
      bfr[nb].q[1] = *(const u32x4*)(wp + 16);
    }
    #pragma unroll
    for (int nb = 0; nb < 4; ++nb)              // ...then a dense WMMA burst
      acc[nb] = wmma_bf16(a, bfr[nb], acc[nb]);
  }

  // Epilogue: C layout — VGPR r: M = (lane>>4)*8 + r, N = lane&15.
  const int rbase = m0 + wave * 16 + (lane >> 4) * 8;
  #pragma unroll
  for (int nb = 0; nb < 4; ++nb) {
    const int n = n0 + nb * 16 + lcol;
    const float bv = bias[n];
    const int h = n >> 6, dh = n & 63;
    #pragma unroll
    for (int r = 0; r < 8; ++r) {
      const int m = rbase + r;
      const int bi = m >> 10, s = m & 1023;     // S = 1024
      Out[(((size_t)(bi * H + h)) * S + s) * 64 + dh] = (__bf16)(acc[nb][r] + bv);
    }
  }
}

// ---------------------------------------------------------------------------
// Kernel 2: flash attention per (b, h, 128 q-rows). Wave w owns 16 q-rows.
// 64-key chunks; K/V chunk staged via registers (pipelined with compute).
// ---------------------------------------------------------------------------
__global__ __launch_bounds__(256) void k_attn(
    const __bf16* __restrict__ Q, const __bf16* __restrict__ K,
    const __bf16* __restrict__ V, const int* __restrict__ mask,
    __bf16* __restrict__ Ctx) {
  constexpr int S = 1024, H = 16, Dh = 64, D = 1024;
  __shared__ __bf16 Ks[64][72];      // [key][dh]   (natural)
  __shared__ __bf16 Vt[64][72];      // [dh][key]   (transposed)
  __shared__ __bf16 Pp[8][16][72];   // per-wave P tile [qrow][key]

  const int tid = threadIdx.x, lane = tid & 31, wave = tid >> 5;
  const int q0 = blockIdx.x * 128;
  const int h = blockIdx.y, b = blockIdx.z;
  const size_t headBase = ((size_t)(b * H + h)) * S * Dh;

  const int lcol = lane & 15;
  const int kb   = (lane >> 4) * 8;
  const int bk16 = (lane >> 4) * 8;
  const int qrow = q0 + wave * 16 + lcol;            // A-frag row
  const int srow = q0 + wave * 16 + (lane >> 4) * 8; // C-layout row base (+r)

  // Preload Q fragments straight from global (row-contiguous bf16).
  FragBF aq[2];
  {
    const __bf16* qp = Q + headBase + (size_t)qrow * Dh;
    #pragma unroll
    for (int c = 0; c < 2; ++c) {
      aq[c].q[0] = *(const u32x4*)(qp + c * 32 + kb);
      aq[c].q[1] = *(const u32x4*)(qp + c * 32 + 16 + kb);
    }
  }

  // K/V staging: thread -> (key, 16-wide dh slice)
  const int ldKey = tid >> 2, ldPart = (tid & 3) * 16;
  RegTile kreg, vreg;
  auto load_chunk = [&](int n0c) {
    const __bf16* ks = K + headBase + (size_t)(n0c + ldKey) * Dh + ldPart;
    kreg.q[0] = *(const u32x4*)ks;
    kreg.q[1] = *(const u32x4*)(ks + 8);
    const __bf16* vs = V + headBase + (size_t)(n0c + ldKey) * Dh + ldPart;
    vreg.q[0] = *(const u32x4*)vs;
    vreg.q[1] = *(const u32x4*)(vs + 8);
  };
  auto store_chunk = [&]() {
    *(u32x4*)&Ks[ldKey][ldPart]     = kreg.q[0];
    *(u32x4*)&Ks[ldKey][ldPart + 8] = kreg.q[1];
    #pragma unroll
    for (int j = 0; j < 16; ++j) Vt[ldPart + j][ldKey] = vreg.h[j];
  };

  float mrun[8], lrun[8];
  #pragma unroll
  for (int r = 0; r < 8; ++r) { mrun[r] = -3.0e38f; lrun[r] = 0.0f; }
  v8f oacc[4] = {};

  load_chunk(0);
  for (int n0c = 0; n0c < S; n0c += 64) {
    __syncthreads();               // previous chunk's PV reads complete
    store_chunk();
    __syncthreads();
    if (n0c + 64 < S) load_chunk(n0c + 64);  // pipeline next K/V chunk

    // --- scores: S = Q * K^T (contract Dh=64 -> 2 WMMAs per block) ---
    v8f sacc[4] = {};
    #pragma unroll
    for (int c = 0; c < 2; ++c) {
      FragBF bfr[4];
      #pragma unroll
      for (int nb = 0; nb < 4; ++nb) {
        const __bf16* kp = &Ks[nb * 16 + lcol][c * 32 + bk16];
        bfr[nb].q[0] = *(const u32x4*)kp;
        bfr[nb].q[1] = *(const u32x4*)(kp + 16);
      }
      #pragma unroll
      for (int nb = 0; nb < 4; ++nb)
        sacc[nb] = wmma_bf16(aq[c], bfr[nb], sacc[nb]);
    }

    // --- scale + additive mask + per-row chunk max ---
    float bm[8];
    #pragma unroll
    for (int r = 0; r < 8; ++r) bm[r] = -3.0e38f;
    #pragma unroll
    for (int nb = 0; nb < 4; ++nb) {
      const int key = n0c + nb * 16 + lcol;
      const int* mrow = mask + ((size_t)b * S + srow) * S + key;
      #pragma unroll
      for (int r = 0; r < 8; ++r) {
        float s = sacc[nb][r] * 0.125f;                // 1/sqrt(64)
        s += (1.0f - (float)mrow[(size_t)r * S]) * -10000.0f;
        sacc[nb][r] = s;
        bm[r] = fmaxf(bm[r], s);
      }
    }
    #pragma unroll
    for (int r = 0; r < 8; ++r) {
      #pragma unroll
      for (int off = 1; off <= 8; off <<= 1)           // reduce across 16 lanes
        bm[r] = fmaxf(bm[r], __shfl_xor(bm[r], off, 32));
    }

    // --- online softmax update ---
    float alpha[8], rsum[8];
    #pragma unroll
    for (int r = 0; r < 8; ++r) {
      const float mnew = fmaxf(mrun[r], bm[r]);
      alpha[r] = exp2f((mrun[r] - mnew) * LOG2E);
      mrun[r] = mnew;
      rsum[r] = 0.0f;
    }
    #pragma unroll
    for (int nb = 0; nb < 4; ++nb)
      #pragma unroll
      for (int r = 0; r < 8; ++r) oacc[nb][r] *= alpha[r];

    #pragma unroll
    for (int nb = 0; nb < 4; ++nb) {
      #pragma unroll
      for (int r = 0; r < 8; ++r) {
        const float p = exp2f((sacc[nb][r] - mrun[r]) * LOG2E);
        rsum[r] += p;
        Pp[wave][(lane >> 4) * 8 + r][nb * 16 + lcol] = (__bf16)p;
      }
    }
    #pragma unroll
    for (int r = 0; r < 8; ++r) {
      #pragma unroll
      for (int off = 1; off <= 8; off <<= 1)
        rsum[r] += __shfl_xor(rsum[r], off, 32);
      lrun[r] = lrun[r] * alpha[r] + rsum[r];
    }
    __syncthreads();  // P visible for refragmentation

    // --- PV: O += P * V (contract keys, 32 per WMMA) ---
    #pragma unroll
    for (int kkc = 0; kkc < 2; ++kkc) {
      FragBF pa;
      const __bf16* pp = &Pp[wave][lcol][kkc * 32 + kb];
      pa.q[0] = *(const u32x4*)pp;
      pa.q[1] = *(const u32x4*)(pp + 16);
      FragBF vb[4];
      #pragma unroll
      for (int nb = 0; nb < 4; ++nb) {
        const __bf16* vp = &Vt[nb * 16 + lcol][kkc * 32 + bk16];
        vb[nb].q[0] = *(const u32x4*)vp;
        vb[nb].q[1] = *(const u32x4*)(vp + 16);
      }
      #pragma unroll
      for (int nb = 0; nb < 4; ++nb)
        oacc[nb] = wmma_bf16(pa, vb[nb], oacc[nb]);
    }
  }

  // Epilogue: normalize and emit context in [B*S][D] bf16 row-major.
  #pragma unroll
  for (int nb = 0; nb < 4; ++nb) {
    const int dh = nb * 16 + lcol;
    #pragma unroll
    for (int r = 0; r < 8; ++r) {
      const float o = oacc[nb][r] / lrun[r];
      Ctx[((size_t)(b * S + srow + r)) * D + h * 64 + dh] = (__bf16)o;
    }
  }
}

// ---------------------------------------------------------------------------
// Kernel 3: Out(fp32) = Ctx(bf16) * Wo(fp32) + bo. Same pipelined tiling.
// ---------------------------------------------------------------------------
__global__ __launch_bounds__(256) void k_gemm_out(
    const __bf16* __restrict__ A, const float* __restrict__ W,
    const float* __restrict__ bias, float* __restrict__ Out) {
  constexpr int Dk = 1024, N = 1024;
  __shared__ __bf16 As[128][40];
  __shared__ __bf16 Ws[64][40];
  const int tid = threadIdx.x;
  const int lane = tid & 31, wave = tid >> 5;
  const int m0 = blockIdx.y * 128, n0 = blockIdx.x * 64;

  v8f acc[4] = {};

  const int xrow = tid >> 1, xcs = (tid & 1) * 16;
  const int wi = tid * 8, wk = wi >> 6, wn = wi & 63;
  const int ar = wave * 16 + (lane & 15);
  const int kb = (lane >> 4) * 8;
  const int bk = (lane >> 4) * 8;
  const int lcol = lane & 15;

  u32x4 areg[2];
  float wr[8];
  auto load_tiles = [&](int k0) {
    const __bf16* as = A + (size_t)(m0 + xrow) * Dk + k0 + xcs;
    areg[0] = *(const u32x4*)as;
    areg[1] = *(const u32x4*)(as + 8);
    const float* ws = W + (size_t)(k0 + wk) * N + n0 + wn;
    #pragma unroll
    for (int j = 0; j < 8; ++j) wr[j] = ws[j];
  };
  auto store_tiles = [&]() {
    *(u32x4*)&As[xrow][xcs]     = areg[0];
    *(u32x4*)&As[xrow][xcs + 8] = areg[1];
    #pragma unroll
    for (int j = 0; j < 8; ++j) Ws[wn + j][wk] = (__bf16)wr[j];
  };

  load_tiles(0);
  for (int k0 = 0; k0 < Dk; k0 += 32) {
    __syncthreads();
    store_tiles();
    __syncthreads();
    if (k0 + 32 < Dk) load_tiles(k0 + 32);

    FragBF a;
    a.q[0] = *(const u32x4*)&As[ar][kb];
    a.q[1] = *(const u32x4*)&As[ar][16 + kb];
    FragBF bfr[4];
    #pragma unroll
    for (int nb = 0; nb < 4; ++nb) {
      const __bf16* wp = &Ws[nb * 16 + lcol][bk];
      bfr[nb].q[0] = *(const u32x4*)wp;
      bfr[nb].q[1] = *(const u32x4*)(wp + 16);
    }
    #pragma unroll
    for (int nb = 0; nb < 4; ++nb)
      acc[nb] = wmma_bf16(a, bfr[nb], acc[nb]);
  }

  const int rbase = m0 + wave * 16 + (lane >> 4) * 8;
  #pragma unroll
  for (int nb = 0; nb < 4; ++nb) {
    const int n = n0 + nb * 16 + lcol;
    const float bv = bias[n];
    #pragma unroll
    for (int r = 0; r < 8; ++r)
      Out[(size_t)(rbase + r) * N + n] = acc[nb][r] + bv;
  }
}

// ---------------------------------------------------------------------------
extern "C" void kernel_launch(void* const* d_in, const int* in_sizes, int n_in,
                              void* d_out, int out_size, void* d_ws, size_t ws_size,
                              hipStream_t stream) {
  (void)in_sizes; (void)n_in; (void)out_size; (void)ws_size;
  const float* X   = (const float*)d_in[0];
  const int*  mask = (const int*) d_in[1];
  const float* Wq = (const float*)d_in[2];
  const float* bq = (const float*)d_in[3];
  const float* Wk = (const float*)d_in[4];
  const float* bk = (const float*)d_in[5];
  const float* Wv = (const float*)d_in[6];
  const float* bv = (const float*)d_in[7];
  const float* Wo = (const float*)d_in[8];
  const float* bo = (const float*)d_in[9];
  float* out = (float*)d_out;

  // Workspace: Q, K, V in [B,H,S,Dh] bf16; Ctx in [B*S,D] bf16. 4 x 32 MB.
  constexpr size_t ELEMS = (size_t)16 * 1024 * 1024;  // B*S*D
  __bf16* Qb = (__bf16*)d_ws;
  __bf16* Kb = Qb + ELEMS;
  __bf16* Vb = Kb + ELEMS;
  __bf16* Cb = Vb + ELEMS;

  const dim3 blk(256);
  const dim3 gGemm(1024 / 64, 16384 / 128);   // (N tiles, M tiles)
  k_gemm_qkv<<<gGemm, blk, 0, stream>>>(X, Wq, bq, Qb);
  k_gemm_qkv<<<gGemm, blk, 0, stream>>>(X, Wk, bk, Kb);
  k_gemm_qkv<<<gGemm, blk, 0, stream>>>(X, Wv, bv, Vb);

  const dim3 gAttn(1024 / 128, 16, 16);       // (q tiles, H, B)
  k_attn<<<gAttn, blk, 0, stream>>>(Qb, Kb, Vb, mask, Cb);

  k_gemm_out<<<gGemm, blk, 0, stream>>>(Cb, Wo, bo, out);
}
